// EncoderBlock_83837761618015
// MI455X (gfx1250) — matrix operand order repaired and checked by reading, not compiled, not run on hardware
//
#include <hip/hip_runtime.h>
#include <hip/hip_bf16.h>

#define LSEQ   2048
#define NH     16
#define DH     64
#define DM     1024
#define DFF    4096
#define BATCH  2
#define WIN    64

typedef __hip_bfloat16 bf16;
typedef __attribute__((ext_vector_type(16))) __bf16 v16bf;
typedef __attribute__((ext_vector_type(8)))  float  v8f;

union FragU { uint4 u[2]; v16bf v; };

// A-fragment (16x32 bf16, row-major source, ld in elements):
// lanes 0-15: row M=lane, K = 0..15; lanes 16-31: K = 16..31
__device__ __forceinline__ v16bf load_fragA(const bf16* base, int ld, int lane) {
    int r  = lane & 15;
    int ko = (lane >> 4) << 4;            // 0 or 16
    const bf16* p = base + (size_t)r * ld + ko;
    FragU f;
    f.u[0] = *reinterpret_cast<const uint4*>(p);
    f.u[1] = *reinterpret_cast<const uint4*>(p + 8);
    return f.v;
}

// B-fragment (32x16 bf16), source stored [N,K] (col n's K-dim contiguous):
// lanes 0-15: col n, K=0..15; lanes 16-31: col n, K=16..31
__device__ __forceinline__ v16bf load_fragB(const bf16* base, int ld, int lane) {
    int c  = lane & 15;
    int ko = (lane >> 4) << 4;            // 0 or 16
    const bf16* p = base + (size_t)c * ld + ko;
    FragU f;
    f.u[0] = *reinterpret_cast<const uint4*>(p);
    f.u[1] = *reinterpret_cast<const uint4*>(p + 8);
    return f.v;
}

// gfx1250 async global->LDS copy (ASYNCcnt path), 16B per lane.
__device__ __forceinline__ void async_copy_b128(unsigned lds_byte_addr, const void* gaddr) {
    asm volatile("global_load_async_to_lds_b128 %0, %1, off"
                 :: "v"(lds_byte_addr), "v"(gaddr) : "memory");
}

// ---------------------------------------------------------------------------
// bf16 WMMA GEMM, LDS-staged + double-buffered via async-to-LDS.
// Block: 256 thr (8 waves, 2M x 4N), block tile 128x256, wave tile 64x64.
// LDS pitch 40 elems (80B): 16 lanes x 20 banks covers all 64 banks.
// MODE 0: f32 out (+residual)   MODE 1: bf16 out (+exact GELU)
// MODE 2: bf16 out [b,h,l,dh]   MODE 3: bf16 out [b,h,dh,l]
// ---------------------------------------------------------------------------
#define LDP 40

template<int MODE, bool GELU>
__global__ __launch_bounds__(256, 1)
void gemm_bf16_kernel(const bf16* __restrict__ A, const bf16* __restrict__ Bt,
                      const float* __restrict__ bias, const float* __restrict__ res,
                      void* __restrict__ outp, int M, int N, int K)
{
    __shared__ bf16 lA[2][128 * LDP];
    __shared__ bf16 lB[2][256 * LDP];

    const int tid  = threadIdx.x;
    const int lane = tid & 31;
    const int wave = tid >> 5;
    const int m0b  = blockIdx.x * 128;
    const int n0b  = blockIdx.y * 256;
    const int m0w  = (wave & 1) * 64;
    const int n0w  = (wave >> 1) * 64;
    (void)M;

    // per-thread copy slots: A tile 128 rows x 4 chunks (16B); B tile 256 x 4
    const int arow0 = tid >> 1;                 // 2 A-slots/thread
    const int ach0  = (tid & 1) * 2;            // chunks {0,1} or {2,3}
    const int brow0 = tid;                      // 4 B-slots/thread (rows tid/4 + 64*q, chunk tid%4)

    auto issue_stage = [&](int s, int kk) {
#pragma unroll
        for (int q = 0; q < 2; q++) {           // A: 2 async instrs / wave
            int row = arow0, ch = ach0 + q;
            async_copy_b128((unsigned)(size_t)&lA[s][row * LDP + ch * 8],
                            (const void*)(A + (size_t)(m0b + row) * K + kk + ch * 8));
        }
#pragma unroll
        for (int q = 0; q < 4; q++) {           // B: 4 async instrs / wave
            int idx = q * 256 + brow0;
            int row = idx >> 2, ch = idx & 3;
            async_copy_b128((unsigned)(size_t)&lB[s][row * LDP + ch * 8],
                            (const void*)(Bt + (size_t)(n0b + row) * K + kk + ch * 8));
        }
    };

    v8f acc[4][4];
#pragma unroll
    for (int i = 0; i < 4; i++)
#pragma unroll
        for (int j = 0; j < 4; j++) { v8f z = {}; acc[i][j] = z; }

    issue_stage(0, 0);
    int it = 0;
    for (int kk = 0; kk < K; kk += 32, it ^= 1) {
        const bool more = (kk + 32 < K);
        if (more) {
            issue_stage(it ^ 1, kk + 32);
            asm volatile("s_wait_asynccnt 0x6" ::: "memory");  // current stage done
        } else {
            asm volatile("s_wait_asynccnt 0x0" ::: "memory");
        }
        __syncthreads();

        v16bf af[4], bfr[4];
#pragma unroll
        for (int i = 0; i < 4; i++)
            af[i] = load_fragA(&lA[it][(m0w + 16 * i) * LDP], LDP, lane);
#pragma unroll
        for (int j = 0; j < 4; j++)
            bfr[j] = load_fragB(&lB[it][(n0w + 16 * j) * LDP], LDP, lane);
#pragma unroll
        for (int i = 0; i < 4; i++)
#pragma unroll
            for (int j = 0; j < 4; j++)
                acc[i][j] = __builtin_amdgcn_wmma_f32_16x16x32_bf16(
                    false, af[i], false, bfr[j], (short)0, acc[i][j], false, false);
        __syncthreads();   // all waves done with stage `it` before its next overwrite
    }

    const int col = lane & 15;
    const int rb  = (lane >> 4) << 3;     // C layout: lanes>=16 hold rows 8..15
#pragma unroll
    for (int i = 0; i < 4; i++)
#pragma unroll
        for (int j = 0; j < 4; j++)
#pragma unroll
            for (int r = 0; r < 8; r++) {
                int gm = m0b + m0w + 16 * i + rb + r;
                int gn = n0b + n0w + 16 * j + col;
                float v = acc[i][j][r] + bias[gn];
                if (MODE == 0) {
                    if (res) v += res[(size_t)gm * N + gn];
                    ((float*)outp)[(size_t)gm * N + gn] = v;
                } else if (MODE == 1) {
                    if (GELU) v = 0.5f * v * (1.0f + erff(v * 0.70710678118654752f));
                    ((bf16*)outp)[(size_t)gm * N + gn] = (bf16)v;
                } else if (MODE == 2) {
                    int b = gm / LSEQ, l = gm % LSEQ, hh = gn >> 6, d = gn & 63;
                    ((bf16*)outp)[(((size_t)(b * NH + hh)) * LSEQ + l) * DH + d] = (bf16)v;
                } else {
                    int b = gm / LSEQ, l = gm % LSEQ, hh = gn >> 6, d = gn & 63;
                    ((bf16*)outp)[(((size_t)(b * NH + hh)) * DH + d) * LSEQ + l] = (bf16)v;
                }
            }
}

// ---------------------------------------------------------------------------
// Sliding-window ALiBi attention: one wave per (b, h, 16-query block).
// ---------------------------------------------------------------------------
__global__ __launch_bounds__(32, 1)
void attn_kernel(const bf16* __restrict__ q, const bf16* __restrict__ k,
                 const bf16* __restrict__ vt, bf16* __restrict__ ctx)
{
    __shared__ bf16 P[16 * 160];
    const int lane = threadIdx.x;
    const int qb = blockIdx.x & (LSEQ / 16 - 1);
    const int bh = blockIdx.x >> 7;                 // LSEQ/16 == 128
    const int h  = bh & (NH - 1);
    const int b  = bh >> 4;

    const bf16* qp = q  + (size_t)bh * LSEQ * DH;
    const bf16* kp = k  + (size_t)bh * LSEQ * DH;
    const bf16* vp = vt + (size_t)bh * DH * LSEQ;

    v16bf qf0 = load_fragA(qp + (size_t)qb * 16 * DH, DH, lane);
    v16bf qf1 = load_fragA(qp + (size_t)qb * 16 * DH + 32, DH, lane);

    v8f acc[9];
#pragma unroll
    for (int t = 0; t < 9; t++) { v8f z = {}; acc[t] = z; }

    const int kb0 = qb - 4;
#pragma unroll
    for (int t = 0; t < 9; t++) {
        int kb = kb0 + t;
        if (kb < 0 || kb >= LSEQ / 16) continue;
        v16bf k0 = load_fragB(kp + (size_t)kb * 16 * DH, DH, lane);
        v16bf k1 = load_fragB(kp + (size_t)kb * 16 * DH + 32, DH, lane);
        acc[t] = __builtin_amdgcn_wmma_f32_16x16x32_bf16(false, qf0, false, k0, (short)0, acc[t], false, false);
        acc[t] = __builtin_amdgcn_wmma_f32_16x16x32_bf16(false, qf1, false, k1, (short)0, acc[t], false, false);
    }

    const float slope = exp2f(-0.5f * (float)(h + 1));   // 2^(-8*(h+1)/16)
    const int col = lane & 15;
    const int rb  = (lane >> 4) << 3;

#pragma unroll
    for (int t = 0; t < 9; t++)
#pragma unroll
        for (int r = 0; r < 8; r++) {
            int qi = qb * 16 + rb + r;
            int kj = (kb0 + t) * 16 + col;
            int rel = qi - kj;
            int arel = rel < 0 ? -rel : rel;
            float s = acc[t][r] * 0.125f - slope * (float)arel;  // 1/sqrt(64)
            if (kj < 0 || kj >= LSEQ || arel > WIN) s = -1e9f;
            acc[t][r] = s;
        }

    float mx[8], inv[8];
#pragma unroll
    for (int r = 0; r < 8; r++) {
        float m = acc[0][r];
#pragma unroll
        for (int t = 1; t < 9; t++) m = fmaxf(m, acc[t][r]);
        for (int off = 1; off < 16; off <<= 1) m = fmaxf(m, __shfl_xor(m, off, 32));
        mx[r] = m;
    }
#pragma unroll
    for (int r = 0; r < 8; r++) {
        float s = 0.f;
#pragma unroll
        for (int t = 0; t < 9; t++) {
            float e = __expf(acc[t][r] - mx[r]);
            acc[t][r] = e; s += e;
        }
        for (int off = 1; off < 16; off <<= 1) s += __shfl_xor(s, off, 32);
        inv[r] = 1.f / s;
    }
#pragma unroll
    for (int t = 0; t < 9; t++)
#pragma unroll
        for (int r = 0; r < 8; r++)
            P[(rb + r) * 160 + t * 16 + col] = (bf16)(acc[t][r] * inv[r]);
    for (int idx = lane; idx < 256; idx += 32)          // zero K-pad cols 144..159
        P[(idx & 15) * 160 + 144 + (idx >> 4)] = (bf16)0.f;
    __syncthreads();

    v8f cacc[4];
#pragma unroll
    for (int j = 0; j < 4; j++) { v8f z = {}; cacc[j] = z; }

    const int j0 = kb0 * 16;
#pragma unroll
    for (int kc = 0; kc < 5; kc++) {
        int kb = kc * 32;
        v16bf pf = load_fragA(&P[0] + kb, 160, lane);
        int c  = lane & 15;
        int ko = (lane >> 4) << 4;
        int key = j0 + kb + ko;                 // aligned 16-key span
        bool kvalid = (key >= 0) && (key < LSEQ);
#pragma unroll
        for (int j = 0; j < 4; j++) {
            FragU f;
            if (kvalid) {
                const bf16* p = vp + (size_t)(j * 16 + c) * LSEQ + key;
                f.u[0] = *reinterpret_cast<const uint4*>(p);
                f.u[1] = *reinterpret_cast<const uint4*>(p + 8);
            } else {
                f.u[0] = make_uint4(0, 0, 0, 0);
                f.u[1] = make_uint4(0, 0, 0, 0);
            }
            cacc[j] = __builtin_amdgcn_wmma_f32_16x16x32_bf16(false, pf, false, f.v, (short)0, cacc[j], false, false);
        }
    }

    const size_t tokbase = (size_t)b * LSEQ + (size_t)qb * 16;
#pragma unroll
    for (int j = 0; j < 4; j++)
#pragma unroll
        for (int r = 0; r < 8; r++)
            ctx[(tokbase + rb + r) * DM + h * DH + j * 16 + col] = (bf16)cacc[j][r];
}

// ---------------------------------------------------------------------------
// LayerNorm over D=1024 per row; writes f32 and optional bf16 copy.
// ---------------------------------------------------------------------------
__global__ __launch_bounds__(256, 1)
void ln_kernel(const float* __restrict__ y, const float* __restrict__ g,
               const float* __restrict__ be, float* __restrict__ outf,
               bf16* __restrict__ outb)
{
    __shared__ float red[2][8];
    const int row = blockIdx.x;
    const float* yr = y + (size_t)row * DM;
    float vals[4], s = 0.f, ss = 0.f;
#pragma unroll
    for (int i = 0; i < 4; i++) {
        float v = yr[threadIdx.x + i * 256];
        vals[i] = v; s += v; ss += v * v;
    }
    for (int off = 1; off < 32; off <<= 1) {
        s  += __shfl_xor(s,  off, 32);
        ss += __shfl_xor(ss, off, 32);
    }
    if ((threadIdx.x & 31) == 0) { red[0][threadIdx.x >> 5] = s; red[1][threadIdx.x >> 5] = ss; }
    __syncthreads();
    s = 0.f; ss = 0.f;
#pragma unroll
    for (int w = 0; w < 8; w++) { s += red[0][w]; ss += red[1][w]; }
    float mean = s * (1.f / DM);
    float var  = ss * (1.f / DM) - mean * mean;
    float rstd = rsqrtf(var + 1e-5f);
#pragma unroll
    for (int i = 0; i < 4; i++) {
        int n = threadIdx.x + i * 256;
        float o = (vals[i] - mean) * rstd * g[n] + be[n];
        outf[(size_t)row * DM + n] = o;
        if (outb) outb[(size_t)row * DM + n] = (bf16)o;
    }
}

__global__ __launch_bounds__(256, 1)
void f32_to_bf16_kernel(const float* __restrict__ in, bf16* __restrict__ out, int n)
{
    int i = blockIdx.x * 256 + threadIdx.x;
    if (i < n) out[i] = (bf16)in[i];
}

// W [K,N] f32  ->  Wt [N,K] bf16 (32x32 LDS tiles)
__global__ __launch_bounds__(256, 1)
void transpose_to_bf16_kernel(const float* __restrict__ in, bf16* __restrict__ out,
                              int K, int N)
{
    __shared__ float tile[32][33];
    int kb = blockIdx.x * 32, nb = blockIdx.y * 32;
    int tx = threadIdx.x & 31, ty = threadIdx.x >> 5;
    for (int i = ty; i < 32; i += 8)
        tile[i][tx] = in[(size_t)(kb + i) * N + nb + tx];
    __syncthreads();
    for (int i = ty; i < 32; i += 8)
        out[(size_t)(nb + i) * K + kb + tx] = (bf16)tile[tx][i];
}

extern "C" void kernel_launch(void* const* d_in, const int* in_sizes, int n_in,
                              void* d_out, int out_size, void* d_ws, size_t ws_size,
                              hipStream_t stream)
{
    (void)in_sizes; (void)n_in; (void)out_size; (void)ws_size;
    const size_t NTOK = (size_t)BATCH * LSEQ;   // 4096

    size_t off = 0;
    auto carve = [&](size_t bytes) -> void* {
        void* p = (char*)d_ws + off;
        off += (bytes + 255) & ~(size_t)255;
        return p;
    };
    bf16*  xb   = (bf16*)carve(NTOK * DM * 2);
    bf16*  wqt  = (bf16*)carve((size_t)DM * DM * 2);
    bf16*  wkt  = (bf16*)carve((size_t)DM * DM * 2);
    bf16*  wvt  = (bf16*)carve((size_t)DM * DM * 2);
    bf16*  wot  = (bf16*)carve((size_t)DM * DM * 2);
    bf16*  w1t  = (bf16*)carve((size_t)DM * DFF * 2);
    bf16*  w2t  = (bf16*)carve((size_t)DM * DFF * 2);
    bf16*  qbuf = (bf16*)carve(NTOK * DM * 2);
    bf16*  kbuf = (bf16*)carve(NTOK * DM * 2);
    bf16*  vtb  = (bf16*)carve(NTOK * DM * 2);
    bf16*  ctxb = (bf16*)carve(NTOK * DM * 2);
    float* y1   = (float*)carve(NTOK * DM * 4);
    float* hf   = (float*)carve(NTOK * DM * 4);
    bf16*  hb   = (bf16*)carve(NTOK * DM * 2);
    bf16*  ff1  = (bf16*)carve(NTOK * DFF * 2);
    float* y2   = (float*)carve(NTOK * DM * 4);

    const float* x  = (const float*)d_in[0];
    const float* Wq = (const float*)d_in[1];  const float* bq = (const float*)d_in[2];
    const float* Wk = (const float*)d_in[3];  const float* bk = (const float*)d_in[4];
    const float* Wv = (const float*)d_in[5];  const float* bv = (const float*)d_in[6];
    const float* Wo = (const float*)d_in[7];  const float* bo = (const float*)d_in[8];
    const float* W1 = (const float*)d_in[9];  const float* b1 = (const float*)d_in[10];
    const float* W2 = (const float*)d_in[11]; const float* b2 = (const float*)d_in[12];
    const float* g1 = (const float*)d_in[13]; const float* be1 = (const float*)d_in[14];
    const float* g2 = (const float*)d_in[15]; const float* be2 = (const float*)d_in[16];
    float* out = (float*)d_out;

    // 1. precision conversion / weight transposes
    {
        int n = (int)(NTOK * DM);
        f32_to_bf16_kernel<<<n / 256, 256, 0, stream>>>(x, xb, n);
    }
    transpose_to_bf16_kernel<<<dim3(DM / 32, DM / 32), 256, 0, stream>>>(Wq, wqt, DM, DM);
    transpose_to_bf16_kernel<<<dim3(DM / 32, DM / 32), 256, 0, stream>>>(Wk, wkt, DM, DM);
    transpose_to_bf16_kernel<<<dim3(DM / 32, DM / 32), 256, 0, stream>>>(Wv, wvt, DM, DM);
    transpose_to_bf16_kernel<<<dim3(DM / 32, DM / 32), 256, 0, stream>>>(Wo, wot, DM, DM);
    transpose_to_bf16_kernel<<<dim3(DM / 32, DFF / 32), 256, 0, stream>>>(W1, w1t, DM, DFF);
    transpose_to_bf16_kernel<<<dim3(DFF / 32, DM / 32), 256, 0, stream>>>(W2, w2t, DFF, DM);

    // 2. QKV projections (q/k in head layout, v transposed per head)
    dim3 gQ((unsigned)(NTOK / 128), DM / 256);
    gemm_bf16_kernel<2, false><<<gQ, 256, 0, stream>>>(xb, wqt, bq, nullptr, qbuf, (int)NTOK, DM, DM);
    gemm_bf16_kernel<2, false><<<gQ, 256, 0, stream>>>(xb, wkt, bk, nullptr, kbuf, (int)NTOK, DM, DM);
    gemm_bf16_kernel<3, false><<<gQ, 256, 0, stream>>>(xb, wvt, bv, nullptr, vtb,  (int)NTOK, DM, DM);

    // 3. sliding-window ALiBi attention
    attn_kernel<<<BATCH * NH * (LSEQ / 16), 32, 0, stream>>>(qbuf, kbuf, vtb, ctxb);

    // 4. output projection + residual (y1 = x + ctx @ Wo + bo)
    gemm_bf16_kernel<0, false><<<gQ, 256, 0, stream>>>(ctxb, wot, bo, x, y1, (int)NTOK, DM, DM);

    // 5. LN1 -> h (f32 + bf16)
    ln_kernel<<<(unsigned)NTOK, 256, 0, stream>>>(y1, g1, be1, hf, hb);

    // 6. FFN1 with exact GELU (bf16 out)
    dim3 gF1((unsigned)(NTOK / 128), DFF / 256);
    gemm_bf16_kernel<1, true><<<gF1, 256, 0, stream>>>(hb, w1t, b1, nullptr, ff1, (int)NTOK, DFF, DM);

    // 7. FFN2 + residual (y2 = h + ff1 @ W2 + b2)
    gemm_bf16_kernel<0, false><<<gQ, 256, 0, stream>>>(ff1, w2t, b2, hf, y2, (int)NTOK, DM, DFF);

    // 8. LN2 -> final output
    ln_kernel<<<(unsigned)NTOK, 256, 0, stream>>>(y2, g2, be2, out, nullptr);
}
